// NoisyTopKGating_90855738179655
// MI455X (gfx1250) — matrix operand order repaired and checked by reading, not compiled, not run on hardware
//
#include <hip/hip_runtime.h>
#include <hip/hip_bf16.h>
#include <math.h>

// ---- shapes (fixed by the reference) ----
#define B_ROWS 32768
#define DDIM   2048
#define EDIM   64
#define TOPK   8
#define MBLK   128             // rows per workgroup
#define KC     32              // K-chunk == WMMA bf16 K
#define NTHR   256             // 8 waves
#define NITER  (DDIM / KC)     // 64

typedef __attribute__((ext_vector_type(16))) __bf16   v16bf;
typedef __attribute__((ext_vector_type(8)))  float    v8f;
typedef __attribute__((ext_vector_type(4)))  unsigned uint4v;
typedef __attribute__((ext_vector_type(2)))  float    float2v;

union FragU {
    uint4v q[2];   // 2 x 16B LDS loads
    v16bf  v;      // WMMA operand (8 VGPRs)
};

__device__ __forceinline__ void bsplit(float f, unsigned short& h, unsigned short& l) {
    __bf16 hb = (__bf16)f;                 // RNE f32->bf16
    __bf16 lb = (__bf16)(f - (float)hb);   // residual
    h = __builtin_bit_cast(unsigned short, hb);
    l = __builtin_bit_cast(unsigned short, lb);
}

__device__ __forceinline__ void splitpack(float f0, float f1, unsigned& hp, unsigned& lp) {
    unsigned short h0, l0, h1, l1;
    bsplit(f0, h0, l0);
    bsplit(f1, h1, l1);
    hp = (unsigned)h0 | ((unsigned)h1 << 16);
    lp = (unsigned)l0 | ((unsigned)l1 << 16);
}

__device__ __forceinline__ v8f wmma_bf16(v16bf a, v16bf b, v8f c) {
    // (neg_a, A, neg_b, B, c_mod, C, reuse_a, reuse_b)
    return __builtin_amdgcn_wmma_f32_16x16x32_bf16(false, a, false, b, (short)0, c, false, false);
}

__device__ __forceinline__ float softplus_f(float x) {
    return fmaxf(x, 0.0f) + log1pf(expf(-fabsf(x)));
}

// LDS layout in u32 words (16384 u32 = 64 KB total, well under 320 KB/WGP):
//   double-buffered staging, buffer b at u32 offset b*8192:
//     xs_hi +0..2047   xs_lo +2048..4095        128 rows x 16 k-pairs (split bf16)
//     wg_hi +4096..5119  wg_lo +5120..6143      64 cols x 16 k-pairs (n-major)
//     wn_hi +6144..7167  wn_lo +7168..8191
//   epilogue (after k-loop; overlaps both buffers):
//     clean logits f32 [0,8192)   noisy logits f32 [8192,16384)
__global__ __launch_bounds__(NTHR) void noisy_topk_gating_kernel(
    const float* __restrict__ x,
    const float* __restrict__ wg,
    const float* __restrict__ wn,
    const float* __restrict__ noise,
    float* __restrict__ out_w,
    int*   __restrict__ out_idx,
    float* __restrict__ out_clean)
{
    __shared__ uint4v smem4[4096];                 // 64 KB
    unsigned* smem = (unsigned*)smem4;
    const uint4v* s4 = (const uint4v*)smem4;

    const int tid      = threadIdx.x;
    const int wave     = tid >> 5;
    const int lane     = tid & 31;
    const int hs       = lane >> 4;                // lane half-select (0/1)
    const int blockRow = blockIdx.x * MBLK;

    // per-thread staging coordinates (fixed across iterations)
    // x: 8 pairs/thread; pair pi = tid + p*256 -> row pi/16, k-pair pi%16
    // w: 4 pairs/thread/matrix; pi = tid + p*256 -> e = pi%64, k-pair pi/64
    v8f accC[4], accN[4];
    const v8f zz = {0.f, 0.f, 0.f, 0.f, 0.f, 0.f, 0.f, 0.f};
    #pragma unroll
    for (int t = 0; t < 4; ++t) { accC[t] = zz; accN[t] = zz; }

    // ---------------- prologue: stage chunk 0 into buffer 0 ----------------
    {
        #pragma unroll
        for (int p = 0; p < 8; ++p) {
            int pi = tid + (p << 8);
            int r = pi >> 4, k2 = pi & 15;
            const float2v xv = *(const float2v*)(x + (size_t)(blockRow + r) * DDIM + (k2 << 1));
            unsigned hp, lp;
            splitpack(xv.x, xv.y, hp, lp);
            smem[pi] = hp;  smem[2048 + pi] = lp;
        }
        #pragma unroll
        for (int p = 0; p < 4; ++p) {
            int pi = tid + (p << 8);
            int e = pi & 63, k2 = pi >> 6;
            size_t g0 = (size_t)(k2 << 1) * EDIM + e;
            unsigned hp, lp;
            splitpack(wg[g0], wg[g0 + EDIM], hp, lp);
            smem[4096 + e * 16 + k2] = hp;  smem[5120 + e * 16 + k2] = lp;
            splitpack(wn[g0], wn[g0 + EDIM], hp, lp);
            smem[6144 + e * 16 + k2] = hp;  smem[7168 + e * 16 + k2] = lp;
        }
    }
    __syncthreads();

    // ---------------- pipelined main loop: 1 barrier / iteration ----------------
    for (int i = 0; i < NITER; ++i) {
        const int  curq     = (i & 1) << 11;         // uint4 base of current buffer
        const int  nxtu     = ((i + 1) & 1) << 13;   // u32 base of next buffer
        const int  kcn      = (i + 1) * KC;
        const bool haveNext = (i + 1) < NITER;

        // (1) issue global loads for next chunk (in flight during WMMA below)
        float2v xr[8];
        float   wgr[8], wnr[8];
        if (haveNext) {
            #pragma unroll
            for (int p = 0; p < 8; ++p) {
                int pi = tid + (p << 8);
                int r = pi >> 4, k2 = pi & 15;
                xr[p] = *(const float2v*)(x + (size_t)(blockRow + r) * DDIM + kcn + (k2 << 1));
            }
            #pragma unroll
            for (int p = 0; p < 4; ++p) {
                int pi = tid + (p << 8);
                int e = pi & 63, k2 = pi >> 6;
                size_t g0 = (size_t)(kcn + (k2 << 1)) * EDIM + e;
                wgr[2 * p] = wg[g0];  wgr[2 * p + 1] = wg[g0 + EDIM];
                wnr[2 * p] = wn[g0];  wnr[2 * p + 1] = wn[g0 + EDIM];
            }
        }

        // (2) compute on current buffer
        // A fragment (16x32 bf16): lane L -> row L%16; u32 idx row*16 + hs*4 + (v&3) + (v>>2)*8
        FragU Ah, Al;
        const int ab = (wave * 16 + (lane & 15)) * 4 + hs;
        Ah.q[0] = s4[curq + ab];        Ah.q[1] = s4[curq + ab + 2];
        Al.q[0] = s4[curq + 512 + ab];  Al.q[1] = s4[curq + 512 + ab + 2];

        #pragma unroll
        for (int t = 0; t < 4; ++t) {
            // B fragment (32x16 bf16): lane L -> col L%16; u32 idx n*16 + hs*4 + (v&3) + (v>>2)*8
            const int nb = (t * 16 + (lane & 15)) * 4 + hs;
            FragU Bh, Bl;
            Bh.q[0] = s4[curq + 1024 + nb]; Bh.q[1] = s4[curq + 1024 + nb + 2];
            Bl.q[0] = s4[curq + 1280 + nb]; Bl.q[1] = s4[curq + 1280 + nb + 2];
            accC[t] = wmma_bf16(Ah.v, Bh.v, accC[t]);
            accC[t] = wmma_bf16(Ah.v, Bl.v, accC[t]);
            accC[t] = wmma_bf16(Al.v, Bh.v, accC[t]);

            Bh.q[0] = s4[curq + 1536 + nb]; Bh.q[1] = s4[curq + 1536 + nb + 2];
            Bl.q[0] = s4[curq + 1792 + nb]; Bl.q[1] = s4[curq + 1792 + nb + 2];
            accN[t] = wmma_bf16(Ah.v, Bh.v, accN[t]);
            accN[t] = wmma_bf16(Ah.v, Bl.v, accN[t]);
            accN[t] = wmma_bf16(Al.v, Bh.v, accN[t]);
        }

        // (3) convert + store next chunk into the other buffer
        if (haveNext) {
            #pragma unroll
            for (int p = 0; p < 8; ++p) {
                int pi = tid + (p << 8);
                unsigned hp, lp;
                splitpack(xr[p].x, xr[p].y, hp, lp);
                smem[nxtu + pi] = hp;  smem[nxtu + 2048 + pi] = lp;
            }
            #pragma unroll
            for (int p = 0; p < 4; ++p) {
                int pi = tid + (p << 8);
                int e = pi & 63, k2 = pi >> 6;
                unsigned hp, lp;
                splitpack(wgr[2 * p], wgr[2 * p + 1], hp, lp);
                smem[nxtu + 4096 + e * 16 + k2] = hp;  smem[nxtu + 5120 + e * 16 + k2] = lp;
                splitpack(wnr[2 * p], wnr[2 * p + 1], hp, lp);
                smem[nxtu + 6144 + e * 16 + k2] = hp;  smem[nxtu + 7168 + e * 16 + k2] = lp;
            }
        }
        __syncthreads();
    }

    // ---------------- epilogue: accumulators -> LDS logits (reuse both buffers) ----------------
    float* cleanL = (float*)smem4;          // [128][64]
    float* noiseL = cleanL + MBLK * EDIM;   // [128][64]
    #pragma unroll
    for (int t = 0; t < 4; ++t) {
        #pragma unroll
        for (int r = 0; r < 8; ++r) {
            int row = wave * 16 + hs * 8 + r;   // C layout: reg r -> M = r + 8*hs
            int col = t * 16 + (lane & 15);     //           lane -> N = L%16
            cleanL[row * EDIM + col] = accC[t][r];
            noiseL[row * EDIM + col] = accN[t][r];
        }
    }
    __syncthreads();

    // coalesced clean-logits store
    for (int i = tid; i < MBLK * EDIM; i += NTHR)
        out_clean[(size_t)blockRow * EDIM + i] = cleanL[i];

    // per-row noisy logits + top-8 + softmax (1 thread per row)
    if (tid < MBLK) {
        const int row  = tid;
        const size_t g = (size_t)(blockRow + row);

        for (int e = 0; e < EDIM; ++e) {
            float ns = noiseL[row * EDIM + e];
            noiseL[row * EDIM + e] =
                cleanL[row * EDIM + e] + softplus_f(ns) * noise[g * EDIM + e]; // NOISE_STDDEV=1
        }

        unsigned long long used = 0ull;
        float tv[TOPK]; int ti[TOPK];
        #pragma unroll
        for (int j = 0; j < TOPK; ++j) {                 // 8 argmax passes -> sorted desc,
            float best = -__builtin_inff(); int bi = 0;  // first-index tie-break (matches lax.top_k)
            for (int e = 0; e < EDIM; ++e) {
                if ((used >> e) & 1ull) continue;
                float v = noiseL[row * EDIM + e];
                if (v > best) { best = v; bi = e; }
            }
            used |= 1ull << bi;
            tv[j] = best; ti[j] = bi;
        }
        float m = tv[0], s = 0.f, ex[TOPK];
        #pragma unroll
        for (int j = 0; j < TOPK; ++j) { ex[j] = expf(tv[j] - m); s += ex[j]; }
        float inv = 1.f / s;
        #pragma unroll
        for (int j = 0; j < TOPK; ++j) {
            out_w[g * TOPK + j]   = ex[j] * inv;
            out_idx[g * TOPK + j] = ti[j];
        }
    }
}

extern "C" void kernel_launch(void* const* d_in, const int* in_sizes, int n_in,
                              void* d_out, int out_size, void* d_ws, size_t ws_size,
                              hipStream_t stream) {
    const float* x     = (const float*)d_in[0];
    const float* wgt   = (const float*)d_in[1];
    const float* wns   = (const float*)d_in[2];
    const float* noise = (const float*)d_in[3];

    // d_out = concat(combined_weights [B,8] f32, top_k_indices [B,8] i32, clean_logits [B,64] f32)
    float* out_w     = (float*)d_out;
    int*   out_idx   = (int*)((float*)d_out + (size_t)B_ROWS * TOPK);
    float* out_clean = (float*)d_out + 2 * (size_t)B_ROWS * TOPK;

    dim3 grid(B_ROWS / MBLK);   // 256 workgroups
    dim3 block(NTHR);           // 8 waves (wave32)
    noisy_topk_gating_kernel<<<grid, block, 0, stream>>>(x, wgt, wns, noise,
                                                         out_w, out_idx, out_clean);
}
